// GRUModel_8976481648643
// MI455X (gfx1250) — matrix operand order repaired and checked by reading, not compiled, not run on hardware
//
#include <hip/hip_runtime.h>

// ---------------------------------------------------------------------------
// 2-layer GRU (torch GRUCell math, gate order r,z,n) + linear head, MI455X.
// B=1024 batch rows -> 64 workgroups x 16 rows; each WG runs the whole T-loop
// with h0/h1 double-buffered in LDS (bf16). Weights pre-converted to bf16 in
// workspace and streamed from L2 each step (1.2MB << 192MB L2). A zero offset
// is laundered through inline asm once per timestep so the (loop-invariant)
// weight loads can neither be hoisted out of the t-loop (which spilled 40KB
// to scratch) nor lose their global address-space provenance (which degraded
// them to flat_load, tying up the LDS path / DScnt).
// Matrix work: v_wmma_f32_16x16x32_bf16.
// ---------------------------------------------------------------------------

typedef __bf16 bf16_t;
typedef __attribute__((ext_vector_type(16))) __bf16 v16bf;
typedef __attribute__((ext_vector_type(8)))  __bf16 bf16x8;
typedef __attribute__((ext_vector_type(8)))  float  v8f;

#define HID   256
#define I_IN  18
#define B_SZ  1024
#define T_SZ  1024
#define G3H   (3 * HID)          // 768 gate rows
#define HP    264                // padded LDS row stride (bf16), keeps 16B align
#define ROWS  16                 // batch rows per workgroup
#define N1W   (G3H * HID)        // 196608 elements per 768x256 weight

__device__ __forceinline__ v8f vzero8() {
  v8f z;
#pragma unroll
  for (int i = 0; i < 8; i++) z[i] = 0.0f;
  return z;
}

__device__ __forceinline__ float sigmoidf_(float x) {
  return 1.0f / (1.0f + __expf(-x));
}
__device__ __forceinline__ float tanhf_(float x) {
  float e = __expf(2.0f * x);
  return (e - 1.0f) / (e + 1.0f);
}

// B fragment (K=32 x N=16 bf16). Lane l supplies column N=l&15, K-half l>>4:
// 16 contiguous k values => one 32-byte load (2x global_load_b128).
__device__ __forceinline__ v16bf ldB(const bf16_t* base, int rs, int row,
                                     int koff) {
  return *(const v16bf*)(base + (size_t)row * rs + koff);
}

// A fragment (M=16 x K=32 bf16) from LDS. Lane l holds row M=l&15; K-half
// kh=l>>4 gives 16 contiguous k values (two 8-element (16B) chunks), matching B.
__device__ __forceinline__ v16bf ldA(const bf16_t* rowptr, int kc, int kh) {
  union { v16bf v; bf16x8 h[2]; } u;
  u.h[0] = *(const bf16x8*)(rowptr + kc * 32 + 16 * kh);
  u.h[1] = *(const bf16x8*)(rowptr + kc * 32 + 16 * kh + 8);
  return u.v;
}

__device__ __forceinline__ v8f wmma_bf16(v16bf a, v16bf b, v8f c) {
  return __builtin_amdgcn_wmma_f32_16x16x32_bf16(
      /*neg_a=*/false, a, /*neg_b=*/false, b,
      /*c_mod=*/(short)0, c, /*reuse_a=*/false, /*reuse_b=*/false);
}

// One GRU cell evaluation for this wave's 32 output columns.
//   gi = A_x @ Wx^T + bI ; gh = A_h @ Wh^T + bH
//   r = sig(gi_r+gh_r); z = sig(gi_z+gh_z); n = tanh(gi_n + bni + r*(gh_n+bnh))
//   h_next = (1-z)*n + z*h_prev
template <int KXC>
__device__ __forceinline__ void gru_phase(
    const bf16_t* axbuf, int ax_rs,            // LDS A for input GEMM
    const bf16_t* ahbuf,                       // LDS A for hidden GEMM (HP)
    const bf16_t* wx, int wx_rs,               // global bf16 weights (3H x Kx)
    const bf16_t* wh,                          // global bf16 weights (3H x H)
    const float* bI, const float* bH,          // LDS biases (768 each)
    const bf16_t* hprev, bf16_t* hnext,        // LDS h buffers (stride HP)
    int lane, int c0w) {
  const int kh = lane >> 4;
  const int ml = lane & 15;

  v16bf Ax[KXC];
#pragma unroll
  for (int kc = 0; kc < KXC; kc++) Ax[kc] = ldA(axbuf + ml * ax_rs, kc, kh);
  v16bf Ah[8];
#pragma unroll
  for (int kc = 0; kc < 8; kc++) Ah[kc] = ldA(ahbuf + ml * HP, kc, kh);

#pragma unroll
  for (int jt = 0; jt < 2; jt++) {
    const int c0 = c0w + jt * 16;
    const int ncol = c0 + ml;  // this lane's output column (0..255)
    v8f ar = vzero8(), az = vzero8(), ani = vzero8(), anh = vzero8();

#pragma unroll
    for (int kc = 0; kc < KXC; kc++) {
      const int ko = kc * 32 + kh * 16;
      ar  = wmma_bf16(Ax[kc], ldB(wx, wx_rs, 0 * HID + ncol, ko), ar);
      az  = wmma_bf16(Ax[kc], ldB(wx, wx_rs, 1 * HID + ncol, ko), az);
      ani = wmma_bf16(Ax[kc], ldB(wx, wx_rs, 2 * HID + ncol, ko), ani);
    }
#pragma unroll
    for (int kc = 0; kc < 8; kc++) {
      const int ko = kc * 32 + kh * 16;
      ar  = wmma_bf16(Ah[kc], ldB(wh, HID, 0 * HID + ncol, ko), ar);
      az  = wmma_bf16(Ah[kc], ldB(wh, HID, 1 * HID + ncol, ko), az);
      anh = wmma_bf16(Ah[kc], ldB(wh, HID, 2 * HID + ncol, ko), anh);
    }

    const float br  = bI[ncol] + bH[ncol];
    const float bz  = bI[HID + ncol] + bH[HID + ncol];
    const float bni = bI[2 * HID + ncol];
    const float bnh = bH[2 * HID + ncol];

#pragma unroll
    for (int v = 0; v < 8; v++) {
      const int m = v + 8 * kh;  // C layout: VGPR v, lane-half kh -> row
      const float r = sigmoidf_(ar[v] + br);
      const float z = sigmoidf_(az[v] + bz);
      const float n = tanhf_(ani[v] + bni + r * (anh[v] + bnh));
      const float hp = (float)hprev[m * HP + ncol];
      hnext[m * HP + ncol] = (bf16_t)((1.0f - z) * n + z * hp);
    }
  }
}

// Pre-convert weights f32 -> bf16 into workspace (once, L2-resident afterward).
__global__ void conv_weights(const float* __restrict__ wih0f,
                             const float* __restrict__ whh0f,
                             const float* __restrict__ wih1f,
                             const float* __restrict__ whh1f,
                             bf16_t* __restrict__ ws) {
  const int NTOT = 3 * N1W + G3H * 32;
  bf16_t* whh0  = ws;
  bf16_t* wih1  = ws + N1W;
  bf16_t* whh1  = ws + 2 * N1W;
  bf16_t* wih0p = ws + 3 * N1W;  // 768 x 32, zero-padded from 768 x 18
  for (int idx = blockIdx.x * blockDim.x + threadIdx.x; idx < NTOT;
       idx += gridDim.x * blockDim.x) {
    if (idx < N1W) {
      whh0[idx] = (bf16_t)whh0f[idx];
    } else if (idx < 2 * N1W) {
      wih1[idx - N1W] = (bf16_t)wih1f[idx - N1W];
    } else if (idx < 3 * N1W) {
      whh1[idx - 2 * N1W] = (bf16_t)whh1f[idx - 2 * N1W];
    } else {
      const int i = idx - 3 * N1W;
      const int row = i >> 5, c = i & 31;
      wih0p[i] = (bf16_t)((c < I_IN) ? wih0f[row * I_IN + c] : 0.0f);
    }
  }
}

__global__ __launch_bounds__(256) void gru_persistent(
    const float* __restrict__ x,
    const bf16_t* __restrict__ whh0, const bf16_t* __restrict__ wih1,
    const bf16_t* __restrict__ whh1, const bf16_t* __restrict__ wih0p,
    const float* __restrict__ b_ih0, const float* __restrict__ b_hh0,
    const float* __restrict__ b_ih1, const float* __restrict__ b_hh1,
    const float* __restrict__ fc_w, const float* __restrict__ fc_b,
    float* __restrict__ out) {
  __shared__ bf16_t sh_h0[2][ROWS * HP];
  __shared__ bf16_t sh_h1[2][ROWS * HP];
  __shared__ bf16_t sh_x[ROWS * 32];
  __shared__ float  sh_bias[4 * G3H];
  __shared__ float  sh_red[16][17];

  const int tid  = threadIdx.x;
  const int lane = tid & 31;
  const int wave = tid >> 5;
  const int b0   = blockIdx.x * ROWS;
  const int c0w  = wave * 32;  // 8 waves x 32 cols = H

  for (int i = tid; i < G3H; i += 256) {
    sh_bias[i]            = b_ih0[i];
    sh_bias[G3H + i]      = b_hh0[i];
    sh_bias[2 * G3H + i]  = b_ih1[i];
    sh_bias[3 * G3H + i]  = b_hh1[i];
  }
  for (int i = tid; i < ROWS * HP; i += 256) {
    sh_h0[0][i] = (bf16_t)0.0f; sh_h0[1][i] = (bf16_t)0.0f;
    sh_h1[0][i] = (bf16_t)0.0f; sh_h1[1][i] = (bf16_t)0.0f;
  }
  __syncthreads();

  for (int t = 0; t < T_SZ; t++) {
    const int cur = t & 1, nxt = cur ^ 1;

    // Launder a zero byte-offset through inline asm: keeps the weight base
    // pointers' global address-space provenance (=> global_load_b128, not
    // flat_load) while making every weight address loop-variant so the loads
    // cannot be hoisted/CSE'd across timesteps (which previously spilled
    // ~40KB of B-fragments to scratch).
    size_t zoff = 0;
    asm volatile("" : "+s"(zoff));
    const bf16_t* whh0_t = whh0 + zoff;
    const bf16_t* wih0_t = wih0p + zoff;
    const bf16_t* wih1_t = wih1 + zoff;
    const bf16_t* whh1_t = whh1 + zoff;

    // Stage x_t tile (16 rows x 18 -> bf16 padded to 32 cols).
    for (int i = tid; i < ROWS * 32; i += 256) {
      const int r = i >> 5, c = i & 31;
      float v = (c < I_IN)
                    ? x[((size_t)(b0 + r) * T_SZ + t) * I_IN + c]
                    : 0.0f;
      sh_x[(r << 5) + c] = (bf16_t)v;
    }
    if (t + 1 < T_SZ)
      __builtin_prefetch(&x[((size_t)(b0 + (tid & 15)) * T_SZ + t + 1) * I_IN],
                         0, 0);
    __syncthreads();

    // Layer 0: h0_next = GRUCell(x_t, h0_cur)
    gru_phase<1>(sh_x, 32, sh_h0[cur], wih0_t, 32, whh0_t,
                 sh_bias, sh_bias + G3H,
                 sh_h0[cur], sh_h0[nxt], lane, c0w);
    __syncthreads();

    // Layer 1: h1_next = GRUCell(h0_next, h1_cur)
    gru_phase<8>(sh_h0[nxt], HP, sh_h1[cur], wih1_t, HID, whh1_t,
                 sh_bias + 2 * G3H, sh_bias + 3 * G3H,
                 sh_h1[cur], sh_h1[nxt], lane, c0w);
    __syncthreads();
  }

  // Head: out[b] = h1_final . fc_w + fc_b   (h1_final in buffer T&1 == 0)
  const bf16_t* hf = sh_h1[T_SZ & 1];
  {
    const int m = tid >> 4, kp = tid & 15;
    float s = 0.0f;
#pragma unroll
    for (int k = 0; k < 16; k++) {
      const int kk = kp * 16 + k;
      s += (float)hf[m * HP + kk] * fc_w[kk];
    }
    sh_red[m][kp] = s;
  }
  __syncthreads();
  if (tid < ROWS) {
    float s = 0.0f;
#pragma unroll
    for (int kp = 0; kp < 16; kp++) s += sh_red[tid][kp];
    out[b0 + tid] = s + fc_b[0];
  }
}

extern "C" void kernel_launch(void* const* d_in, const int* in_sizes, int n_in,
                              void* d_out, int out_size, void* d_ws,
                              size_t ws_size, hipStream_t stream) {
  const float* x    = (const float*)d_in[0];
  const float* Wih0 = (const float*)d_in[1];
  const float* Whh0 = (const float*)d_in[2];
  const float* bih0 = (const float*)d_in[3];
  const float* bhh0 = (const float*)d_in[4];
  const float* Wih1 = (const float*)d_in[5];
  const float* Whh1 = (const float*)d_in[6];
  const float* bih1 = (const float*)d_in[7];
  const float* bhh1 = (const float*)d_in[8];
  const float* fcw  = (const float*)d_in[9];
  const float* fcb  = (const float*)d_in[10];
  float* out  = (float*)d_out;
  bf16_t* ws  = (bf16_t*)d_ws;  // needs 3*196608 + 768*32 bf16 = ~1.17 MB

  conv_weights<<<512, 256, 0, stream>>>(Wih0, Whh0, Wih1, Whh1, ws);
  gru_persistent<<<B_SZ / ROWS, 256, 0, stream>>>(
      x, ws, ws + N1W, ws + 2 * N1W, ws + 3 * N1W,
      bih0, bhh0, bih1, bhh1, fcw, fcb, out);
}
